// ViT_66709432041786
// MI455X (gfx1250) — matrix operand order repaired and checked by reading, not compiled, not run on hardware
//
#include <hip/hip_runtime.h>
#include <hip/hip_bf16.h>

// ---------------------------------------------------------------------------
// ViT-Base forward for MI455X (gfx1250), bf16 WMMA everywhere.
// Branch-free GEMM inner loop; wave tile 32(M) x 64(N): 8 WMMA per K-step
// sharing 4 B fragments (2x arithmetic density vs 16x64).
// ---------------------------------------------------------------------------

typedef __attribute__((ext_vector_type(16))) __bf16 v16bf;
typedef __attribute__((ext_vector_type(8)))  __bf16 v8bf;
typedef __attribute__((ext_vector_type(8)))  float  v8f;

// Problem dims
constexpr int BATCH   = 32;
constexpr int SEQ     = 197;   // real tokens
constexpr int SP      = 224;   // seq padded to 32 (M and K dims)
constexpr int SNP     = 256;   // seq padded to 64 (N dim for scores)
constexpr int DIM     = 768;
constexpr int NHEAD   = 12;
constexpr int DHEAD   = 64;
constexpr int MLPD    = 3072;
constexpr int NLAYER  = 12;
constexpr int NPATCH  = 196;
constexpr int PK      = 588;   // patch-embed K, = 3*14*14
constexpr int PKP     = 608;   // padded to 32
constexpr int OUTC    = 1000;
constexpr int OUTP    = 1024;  // padded to 64
constexpr int ROWS    = BATCH * SP;      // 7168 (mult of 256)
constexpr int PROWS   = BATCH * NPATCH;  // 6272 (mult of 32)
constexpr int BH      = BATCH * NHEAD;   // 384

// ---------------------------------------------------------------------------
// Generic batched GEMM:  C[z] = A[z] (Mp x Kp, bf16) * Bt[z]^T + bias (+ addC)
// Bt is stored N-major: Bt[n][k] == B[k][n]  (both operand fragments are
// contiguous loads using the same k = 16*lh + s order in A and B).
// Block = 256 threads = 8 waves; wave tile = 32(M) x 64(N); k-step 32.
// Np must be a multiple of 64; Mp a multiple of 32.
// ---------------------------------------------------------------------------
__global__ __launch_bounds__(256) void k_gemm(
    const __bf16* __restrict__ A, const __bf16* __restrict__ Bt,
    float* C, const float* addC,
    const float* __restrict__ bias, int Nbias,
    int Mp, int Kp, int lda, int ldb, int ldc,
    long sA, long sB, int cdiv, long cs0, long cs1, int act_gelu)
{
  const int wave = threadIdx.x >> 5;
  const int lane = threadIdx.x & 31;
  const int ll   = lane & 15;
  const int lh   = lane >> 4;
  const int mBase  = blockIdx.y * 256 + wave * 32;
  if (mBase >= Mp) return;                       // uniform per wave
  const int nBase0 = blockIdx.x * 64;
  const long z = blockIdx.z;

  const __bf16* Az = A + z * sA;
  const __bf16* Bz = Bt + z * sB;
  const long cOff = (z / cdiv) * cs0 + (z % cdiv) * cs1;

  // A fragment source: lane ll = row; lh=0 -> K 0..15, lh=1 -> 16..31
  const __bf16* arow0 = Az + (long)(mBase + ll) * lda + lh * 16;
  const __bf16* arow1 = arow0 + 16L * lda;
  // B fragment source: lane ll = column n; lh selects K halves (16 contiguous)
  const __bf16* brow0 = Bz + (long)(nBase0 + ll) * ldb + lh * 16;
  const long bstep = 16L * ldb;

  v8f acc0[4], acc1[4];
#pragma unroll
  for (int j = 0; j < 4; ++j)
#pragma unroll
    for (int i = 0; i < 8; ++i) { acc0[j][i] = 0.0f; acc1[j][i] = 0.0f; }

  for (int k0 = 0; k0 < Kp; k0 += 32) {
    v8bf a0lo = *(const v8bf*)(arow0 + k0);
    v8bf a0hi = *(const v8bf*)(arow0 + k0 + 8);
    v8bf a1lo = *(const v8bf*)(arow1 + k0);
    v8bf a1hi = *(const v8bf*)(arow1 + k0 + 8);
    v16bf a0, a1;
#pragma unroll
    for (int i = 0; i < 8; ++i) {
      a0[i] = a0lo[i]; a0[8 + i] = a0hi[i];
      a1[i] = a1lo[i]; a1[8 + i] = a1hi[i];
    }

    v16bf b[4];
#pragma unroll
    for (int j = 0; j < 4; ++j)
      b[j] = *(const v16bf*)(brow0 + j * bstep + k0);

#pragma unroll
    for (int j = 0; j < 4; ++j)
      acc0[j] = __builtin_amdgcn_wmma_f32_16x16x32_bf16(
          false, a0, false, b[j], (short)0, acc0[j], false, false);
#pragma unroll
    for (int j = 0; j < 4; ++j)
      acc1[j] = __builtin_amdgcn_wmma_f32_16x16x32_bf16(
          false, a1, false, b[j], (short)0, acc1[j], false, false);
  }

  float* Cz = C + cOff;
  const float* addz = addC ? addC + cOff : nullptr;
#pragma unroll
  for (int half = 0; half < 2; ++half) {
#pragma unroll
    for (int j = 0; j < 4; ++j) {
      const int n  = nBase0 + j * 16 + ll;
      const float bv = (bias != nullptr && n < Nbias) ? bias[n] : 0.0f;
      const int m0 = mBase + half * 16 + lh * 8;
#pragma unroll
      for (int r = 0; r < 8; ++r) {
        const long idx = (long)(m0 + r) * ldc + n;
        float v = (half ? acc1[j][r] : acc0[j][r]) + bv;
        if (addz) v += addz[idx];
        if (act_gelu) v = 0.5f * v * (1.0f + erff(v * 0.70710678118654752f));
        Cz[idx] = v;
      }
    }
  }
}

// ---------------------------------------------------------------------------
// Weight repack kernels (fp32 -> bf16, N-major transpose, zero padding)
// ---------------------------------------------------------------------------
__global__ void k_pack_patchW(const float* __restrict__ W, __bf16* __restrict__ out, long n) {
  long i = (long)blockIdx.x * blockDim.x + threadIdx.x;
  if (i >= n) return;
  int nn = (int)(i / PKP), kk = (int)(i % PKP);
  out[i] = (kk < PK) ? (__bf16)W[(long)kk * DIM + nn] : (__bf16)0.0f;
}

__global__ void k_pack_qkvW(const float* __restrict__ W, __bf16* __restrict__ out, long n) {
  // W: [L][NH][D][DH]  ->  out: [L][n=h*64+e][k=d]
  long i = (long)blockIdx.x * blockDim.x + threadIdx.x;
  if (i >= n) return;
  long l = i / ((long)DIM * DIM);
  long r = i % ((long)DIM * DIM);
  int nn = (int)(r / DIM), kk = (int)(r % DIM);
  int h = nn / DHEAD, e = nn % DHEAD;
  out[i] = (__bf16)W[(((l * NHEAD + h) * DIM + kk) * DHEAD) + e];
}

__global__ void k_pack_T(const float* __restrict__ in, __bf16* __restrict__ out,
                         int K, int N, long n) {
  // in: [L][K][N] -> out: [L][N][K]
  long i = (long)blockIdx.x * blockDim.x + threadIdx.x;
  if (i >= n) return;
  long kn = (long)K * N;
  long l = i / kn, r = i % kn;
  int nn = (int)(r / K), kk = (int)(r % K);
  out[i] = (__bf16)in[l * kn + (long)kk * N + nn];
}

__global__ void k_pack_headW(const float* __restrict__ W, __bf16* __restrict__ out, long n) {
  long i = (long)blockIdx.x * blockDim.x + threadIdx.x;
  if (i >= n) return;
  int nn = (int)(i / DIM), kk = (int)(i % DIM);
  out[i] = (nn < OUTC) ? (__bf16)W[(long)kk * OUTC + nn] : (__bf16)0.0f;
}

// ---------------------------------------------------------------------------
// Activation kernels
// ---------------------------------------------------------------------------
__global__ void k_patches(const float* __restrict__ x, __bf16* __restrict__ out, long n) {
  // out: [b*196+p][k], k = c*256 + ph*16 + pw  (F.unfold order), K-padded
  long i = (long)blockIdx.x * blockDim.x + threadIdx.x;
  if (i >= n) return;
  int row = (int)(i / PKP), kk = (int)(i % PKP);
  if (kk >= PK) { out[i] = (__bf16)0.0f; return; }
  int c = kk >> 8, rem = kk & 255, ph = rem >> 4, pw = rem & 15;
  int b = row / NPATCH, p = row % NPATCH, py = p / 14, px = p % 14;
  out[i] = (__bf16)x[(((long)b * 3 + c) * 224 + (py * 16 + ph)) * 224 + (px * 16 + pw)];
}

__global__ void k_add_cls_pos(const float* __restrict__ tok_tmp,
                              const float* __restrict__ cls,
                              float* __restrict__ tok, long n) {
  long i = (long)blockIdx.x * blockDim.x + threadIdx.x;
  if (i >= n) return;
  int row = (int)(i / DIM), d = (int)(i % DIM);
  int b = row / SP, s = row % SP;
  float v = 0.0f;
  if (s < SEQ) {
    float base = (s == 0) ? cls[d] : tok_tmp[((long)b * NPATCH + (s - 1)) * DIM + d];
    int jj = d & ~1;
    float ang = (float)s * powf(10000.0f, -(float)jj / (float)DIM);
    v = base + ((d & 1) ? cosf(ang) : sinf(ang));
  }
  tok[i] = v;
}

__global__ __launch_bounds__(256) void k_layernorm(
    const float* __restrict__ x, const float* __restrict__ g,
    const float* __restrict__ b, __bf16* __restrict__ hb, float* hf)
{
  const int row = blockIdx.x;
  const int s = row % SP;
  const long base = (long)row * DIM;
  __shared__ float red[256];
  const int tid = threadIdx.x;
  if (s >= SEQ) {
    for (int d = tid; d < DIM; d += 256) {
      hb[base + d] = (__bf16)0.0f;
      if (hf) hf[base + d] = 0.0f;
    }
    return;
  }
  float sum = 0.0f;
  for (int d = tid; d < DIM; d += 256) sum += x[base + d];
  red[tid] = sum; __syncthreads();
  for (int o = 128; o > 0; o >>= 1) { if (tid < o) red[tid] += red[tid + o]; __syncthreads(); }
  const float mean = red[0] / DIM;
  __syncthreads();
  float vs = 0.0f;
  for (int d = tid; d < DIM; d += 256) { float t = x[base + d] - mean; vs += t * t; }
  red[tid] = vs; __syncthreads();
  for (int o = 128; o > 0; o >>= 1) { if (tid < o) red[tid] += red[tid + o]; __syncthreads(); }
  const float inv = rsqrtf(red[0] / DIM + 1e-5f);
  for (int d = tid; d < DIM; d += 256) {
    float y = (x[base + d] - mean) * inv * g[d] + b[d];
    hb[base + d] = (__bf16)y;
    if (hf) hf[base + d] = y;
  }
}

__global__ void k_split_heads(const float* __restrict__ in, __bf16* __restrict__ out,
                              int rowsPad, long n) {
  // in: [b*SP+s][h*64+e] fp32 -> out: [((b*12+h)*rowsPad+s)*64+e] bf16, pad s -> 0
  long i = (long)blockIdx.x * blockDim.x + threadIdx.x;
  if (i >= n) return;
  int e = (int)(i % DHEAD);
  long r = i / DHEAD;
  int s = (int)(r % rowsPad);  r /= rowsPad;
  int h = (int)(r % NHEAD);
  int b = (int)(r / NHEAD);
  out[i] = (s < SEQ) ? (__bf16)in[((long)b * SP + s) * DIM + h * DHEAD + e] : (__bf16)0.0f;
}

__global__ void k_pack_vT(const float* __restrict__ in, __bf16* __restrict__ out, long n) {
  // in: [b*SP+t][h*64+e] fp32 -> out: [((b*12+h)*64+e)*SP+t] bf16, pad t>=SEQ -> 0
  long i = (long)blockIdx.x * blockDim.x + threadIdx.x;
  if (i >= n) return;
  int t = (int)(i % SP);
  long r = i / SP;
  int e = (int)(r % DHEAD); r /= DHEAD;
  int h = (int)(r % NHEAD);
  int b = (int)(r / NHEAD);
  out[i] = (t < SEQ) ? (__bf16)in[((long)b * SP + t) * DIM + h * DHEAD + e] : (__bf16)0.0f;
}

__global__ __launch_bounds__(256) void k_softmax_attn(
    const float* __restrict__ scores, __bf16* __restrict__ attn)
{
  const long row = blockIdx.x;                 // bh*SP + s
  const float* sr = scores + row * SNP;        // scores ldc = SNP
  __bf16* ar = attn + row * SP;
  __shared__ float red[256];
  const int tid = threadIdx.x;
  const float scale = 1.0f / (float)DHEAD;     // source divides by d_head, not sqrt
  float mx = -1e30f;
  for (int t = tid; t < SEQ; t += 256) mx = fmaxf(mx, sr[t] * scale);
  red[tid] = mx; __syncthreads();
  for (int o = 128; o > 0; o >>= 1) { if (tid < o) red[tid] = fmaxf(red[tid], red[tid + o]); __syncthreads(); }
  mx = red[0]; __syncthreads();
  float sum = 0.0f;
  for (int t = tid; t < SEQ; t += 256) sum += expf(sr[t] * scale - mx);
  red[tid] = sum; __syncthreads();
  for (int o = 128; o > 0; o >>= 1) { if (tid < o) red[tid] += red[tid + o]; __syncthreads(); }
  const float inv = 1.0f / red[0];
  for (int t = tid; t < SP; t += 256)
    ar[t] = (t < SEQ) ? (__bf16)(expf(sr[t] * scale - mx) * inv) : (__bf16)0.0f;
}

__global__ void k_mask_rows(const float* __restrict__ in, __bf16* __restrict__ out,
                            int cols, long n) {
  long i = (long)blockIdx.x * blockDim.x + threadIdx.x;
  if (i >= n) return;
  int s = (int)((i / cols) % SP);
  out[i] = (s < SEQ) ? (__bf16)in[i] : (__bf16)0.0f;
}

__global__ void k_pack_cls(const float* __restrict__ tok, __bf16* __restrict__ out, long n) {
  long i = (long)blockIdx.x * blockDim.x + threadIdx.x;
  if (i >= n) return;
  int b = (int)(i / DIM), d = (int)(i % DIM);
  out[i] = (__bf16)tok[(long)b * SP * DIM + d];
}

__global__ __launch_bounds__(256) void k_softmax_out(
    const float* __restrict__ logits, float* __restrict__ out)
{
  const int b = blockIdx.x;
  const float* lr = logits + (long)b * OUTP;
  __shared__ float red[256];
  const int tid = threadIdx.x;
  float mx = -1e30f;
  for (int c = tid; c < OUTC; c += 256) mx = fmaxf(mx, lr[c]);
  red[tid] = mx; __syncthreads();
  for (int o = 128; o > 0; o >>= 1) { if (tid < o) red[tid] = fmaxf(red[tid], red[tid + o]); __syncthreads(); }
  mx = red[0]; __syncthreads();
  float s = 0.0f;
  for (int c = tid; c < OUTC; c += 256) s += expf(lr[c] - mx);
  red[tid] = s; __syncthreads();
  for (int o = 128; o > 0; o >>= 1) { if (tid < o) red[tid] += red[tid + o]; __syncthreads(); }
  const float inv = 1.0f / red[0];
  for (int c = tid; c < OUTC; c += 256) out[(long)b * OUTC + c] = expf(lr[c] - mx) * inv;
}

// ---------------------------------------------------------------------------
// Host launcher
// ---------------------------------------------------------------------------
static inline unsigned nblk(long n) { return (unsigned)((n + 255) / 256); }

extern "C" void kernel_launch(void* const* d_in, const int* in_sizes, int n_in,
                              void* d_out, int out_size, void* d_ws, size_t ws_size,
                              hipStream_t stream) {
  (void)in_sizes; (void)n_in; (void)out_size; (void)ws_size;
  const float* x       = (const float*)d_in[0];
  const float* patch_W = (const float*)d_in[1];
  const float* patch_b = (const float*)d_in[2];
  const float* cls_tok = (const float*)d_in[3];
  const float* ln_g    = (const float*)d_in[4];
  const float* ln_b    = (const float*)d_in[5];
  const float* Wq      = (const float*)d_in[6];
  const float* bq      = (const float*)d_in[7];
  const float* Wk      = (const float*)d_in[8];
  const float* bk      = (const float*)d_in[9];
  const float* Wv      = (const float*)d_in[10];
  const float* bv      = (const float*)d_in[11];
  const float* Wo      = (const float*)d_in[12];
  const float* bo      = (const float*)d_in[13];
  const float* W1      = (const float*)d_in[14];
  const float* b1      = (const float*)d_in[15];
  const float* W2      = (const float*)d_in[16];
  const float* b2      = (const float*)d_in[17];
  const float* head_W  = (const float*)d_in[18];
  const float* head_b  = (const float*)d_in[19];
  float* out = (float*)d_out;

  // ---- workspace bump allocator ----
  char* base = (char*)d_ws;
  size_t off = 0;
  auto alloc = [&](size_t bytes) -> void* {
    void* p = base + off;
    off = (off + bytes + 255) & ~(size_t)255;
    return p;
  };
  const long DD = (long)DIM * DIM;

  __bf16* patchWt = (__bf16*)alloc((size_t)DIM * PKP * 2);
  __bf16* qWt     = (__bf16*)alloc((size_t)NLAYER * DD * 2);
  __bf16* kWt     = (__bf16*)alloc((size_t)NLAYER * DD * 2);
  __bf16* vWt     = (__bf16*)alloc((size_t)NLAYER * DD * 2);
  __bf16* oWt     = (__bf16*)alloc((size_t)NLAYER * DD * 2);
  __bf16* W1t     = (__bf16*)alloc((size_t)NLAYER * MLPD * DIM * 2);
  __bf16* W2t     = (__bf16*)alloc((size_t)NLAYER * DIM * MLPD * 2);
  __bf16* headWt  = (__bf16*)alloc((size_t)OUTP * DIM * 2);

  __bf16* patches = (__bf16*)alloc((size_t)PROWS * PKP * 2);
  __bf16* hB      = (__bf16*)alloc((size_t)ROWS * DIM * 2);
  __bf16* qB      = (__bf16*)alloc((size_t)BH * SP * DHEAD * 2);
  __bf16* kB      = (__bf16*)alloc((size_t)BH * SNP * DHEAD * 2);   // 256 rows (zero-padded)
  __bf16* vT      = (__bf16*)alloc((size_t)BH * DHEAD * SP * 2);
  __bf16* attnB   = (__bf16*)alloc((size_t)BH * SP * SP * 2);
  __bf16* ctxB    = (__bf16*)alloc((size_t)ROWS * DIM * 2);
  __bf16* mlpB    = (__bf16*)alloc((size_t)ROWS * MLPD * 2);
  __bf16* clsB    = (__bf16*)alloc((size_t)BATCH * DIM * 2);

  float* tok    = (float*)alloc((size_t)ROWS * DIM * 4);
  float* ctxF   = (float*)alloc((size_t)ROWS * DIM * 4);
  float* logits = (float*)alloc((size_t)BATCH * OUTP * 4);
  // Time-multiplexed fp32 scratch: tok_tmp / {qf,kf,vf} / scores / mlp
  // ROWS*MLPD == 22,020,096 floats == BH*SP*SNP exactly.
  float* SF = (float*)alloc((size_t)ROWS * MLPD * 4);
  float* qF = SF;
  float* kF = SF + (long)ROWS * DIM;
  float* vF = SF + 2L * ROWS * DIM;
  float* scores = SF;   // valid after q/k/v are consumed
  float* mlpF   = SF;   // valid after scores are consumed
  float* tokTmp = SF;   // patch-embed output, pre-layer

  auto gemm = [&](const __bf16* A, const __bf16* Bt, float* C, const float* addC,
                  const float* bias, int Nbias, int Mp, int Np, int Kp,
                  int lda, int ldb, int ldc, long sA, long sB,
                  int batches, int cdiv, long cs0, long cs1, int gelu) {
    dim3 grid((unsigned)(Np / 64), (unsigned)((Mp + 255) / 256), (unsigned)batches);
    k_gemm<<<grid, 256, 0, stream>>>(A, Bt, C, addC, bias, Nbias, Mp, Kp,
                                     lda, ldb, ldc, sA, sB, cdiv, cs0, cs1, gelu);
  };

  // ---- repack weights to bf16 (N-major) ----
  { long n = (long)DIM * PKP;          k_pack_patchW<<<nblk(n), 256, 0, stream>>>(patch_W, patchWt, n); }
  { long n = (long)NLAYER * DD;        k_pack_qkvW<<<nblk(n), 256, 0, stream>>>(Wq, qWt, n);
                                       k_pack_qkvW<<<nblk(n), 256, 0, stream>>>(Wk, kWt, n);
                                       k_pack_qkvW<<<nblk(n), 256, 0, stream>>>(Wv, vWt, n); }
  { long n = (long)NLAYER * DD;        k_pack_T<<<nblk(n), 256, 0, stream>>>(Wo, oWt, DIM, DIM, n); }
  { long n = (long)NLAYER * DIM * MLPD;
    k_pack_T<<<nblk(n), 256, 0, stream>>>(W1, W1t, DIM, MLPD, n);   // out [L][MLP][D]
    k_pack_T<<<nblk(n), 256, 0, stream>>>(W2, W2t, MLPD, DIM, n); } // out [L][D][MLP]
  { long n = (long)OUTP * DIM;         k_pack_headW<<<nblk(n), 256, 0, stream>>>(head_W, headWt, n); }

  // ---- patch embed + cls + positional ----
  { long n = (long)PROWS * PKP;        k_patches<<<nblk(n), 256, 0, stream>>>(x, patches, n); }
  gemm(patches, patchWt, tokTmp, nullptr, patch_b, DIM,
       PROWS, DIM, PKP, PKP, PKP, DIM, 0, 0, 1, 1, 0, 0, 0);
  { long n = (long)ROWS * DIM;         k_add_cls_pos<<<nblk(n), 256, 0, stream>>>(tokTmp, cls_tok, tok, n); }

  // ---- encoder layers ----
  for (int l = 0; l < NLAYER; ++l) {
    const float* g = ln_g + (long)l * DIM;
    const float* bb = ln_b + (long)l * DIM;

    k_layernorm<<<ROWS, 256, 0, stream>>>(tok, g, bb, hB, nullptr);

    gemm(hB, qWt + (long)l * DD, qF, nullptr, bq + (long)l * DIM, DIM,
         ROWS, DIM, DIM, DIM, DIM, DIM, 0, 0, 1, 1, 0, 0, 0);
    gemm(hB, kWt + (long)l * DD, kF, nullptr, bk + (long)l * DIM, DIM,
         ROWS, DIM, DIM, DIM, DIM, DIM, 0, 0, 1, 1, 0, 0, 0);
    gemm(hB, vWt + (long)l * DD, vF, nullptr, bv + (long)l * DIM, DIM,
         ROWS, DIM, DIM, DIM, DIM, DIM, 0, 0, 1, 1, 0, 0, 0);

    { long n = (long)BH * SP * DHEAD;
      k_split_heads<<<nblk(n), 256, 0, stream>>>(qF, qB, SP, n); }
    { long n = (long)BH * SNP * DHEAD;
      k_split_heads<<<nblk(n), 256, 0, stream>>>(kF, kB, SNP, n); }
    { long n = (long)BH * DHEAD * SP;
      k_pack_vT<<<nblk(n), 256, 0, stream>>>(vF, vT, n); }

    // scores[bh] = q @ k^T   (k stored [t][e] is exactly the N-major operand);
    // N padded to 256 with zero rows of k -> branch-free GEMM, extra cols = 0.
    gemm(qB, kB, scores, nullptr, nullptr, 0,
         SP, SNP, DHEAD, DHEAD, DHEAD, SNP,
         (long)SP * DHEAD, (long)SNP * DHEAD, BH, 1, (long)SP * SNP, 0, 0);

    k_softmax_attn<<<(unsigned)((long)BH * SP), 256, 0, stream>>>(scores, attnB);

    // ctx[bh] = attn @ v  (vT is N-major [e][t]); scatter into [b*SP+s][h*64+e]
    gemm(attnB, vT, ctxF, nullptr, nullptr, 0,
         SP, DHEAD, SP, SP, SP, DIM,
         (long)SP * SP, (long)DHEAD * SP, BH, NHEAD, (long)SP * DIM, DHEAD, 0);

    { long n = (long)ROWS * DIM; k_mask_rows<<<nblk(n), 256, 0, stream>>>(ctxF, ctxB, DIM, n); }

    // tok += ctx @ Wo + bo
    gemm(ctxB, oWt + (long)l * DD, tok, tok, bo + (long)l * DIM, DIM,
         ROWS, DIM, DIM, DIM, DIM, DIM, 0, 0, 1, 1, 0, 0, 0);

    // tok = LN(tok)  (same params reused, per source); also emit bf16 copy
    k_layernorm<<<ROWS, 256, 0, stream>>>(tok, g, bb, hB, tok);

    // mlp = gelu(tok @ W1 + b1)
    gemm(hB, W1t + (long)l * MLPD * DIM, mlpF, nullptr, b1 + (long)l * MLPD, MLPD,
         ROWS, MLPD, DIM, DIM, DIM, MLPD, 0, 0, 1, 1, 0, 0, 1);

    { long n = (long)ROWS * MLPD; k_mask_rows<<<nblk(n), 256, 0, stream>>>(mlpF, mlpB, MLPD, n); }

    // tok += mlp @ W2 + b2
    gemm(mlpB, W2t + (long)l * DIM * MLPD, tok, tok, b2 + (long)l * DIM, DIM,
         ROWS, DIM, MLPD, MLPD, MLPD, DIM, 0, 0, 1, 1, 0, 0, 0);
  }

  // ---- classification head + softmax ----
  { long n = (long)BATCH * DIM; k_pack_cls<<<nblk(n), 256, 0, stream>>>(tok, clsB, n); }
  gemm(clsB, headWt, logits, nullptr, head_b, OUTC,
       BATCH, OUTP, DIM, DIM, DIM, OUTP, 0, 0, 1, 1, 0, 0, 0);
  k_softmax_out<<<BATCH, 256, 0, stream>>>(logits, out);
}
